// GRUSelf_13185549598726
// MI455X (gfx1250) — compile-verified, matrix operand order checked by reading, not run on hardware
//
#include <hip/hip_runtime.h>

// ---------------------------------------------------------------------------
// GRUSelf pipeline for MI455X (gfx1250, wave32, WMMA + TDM).
//   embed -> [WMMA GEMM xp1] -> fused GRU scan (LDS-resident h, WMMA per step)
//   -> [WMMA GEMM whk] -> attention/blend + diffusion noising
//   -> [batched WMMA GEMM W_diff] -> gen_x -> [WMMA GEMM xp2] -> GRU scan 2
//   -> attention pools -> outputs.
// GEMMs: A[M][K] f16 row-major (global, software-pipelined), B[N][K] f16
// staged 64-row panel into LDS via tensor_load_to_lds (TDM), fp32 accumulate
// via v_wmma_f32_16x16x32_f16.
// ---------------------------------------------------------------------------

#define B_   64
#define S_   200
#define C_   32
#define D_   256
#define G3_  768
#define T_   1000
#define SP_  224   // K padded (200 -> 224) for diffusion GEMM
#define MP_  208   // M padded (200 -> 208) for diffusion GEMM
#define BS_  (B_ * S_)

typedef __attribute__((ext_vector_type(16))) _Float16     v16h;
typedef __attribute__((ext_vector_type(8)))  float        v8f;
typedef __attribute__((ext_vector_type(4)))  float        f4;
typedef __attribute__((ext_vector_type(4)))  unsigned int u32x4;
typedef __attribute__((ext_vector_type(8)))  int          i32x8;
typedef __attribute__((ext_vector_type(4)))  int          i32x4;

union Frag {
  v16h     v;
  _Float16 h[16];
  f4       q[2];
};

// A/B fragment (16-bit, 16x32 / 32x16), lane = (row sel nl, half sel hsel):
//   elements 0..7  -> K = kk + hsel*8 + e
//   elements 8..15 -> K = kk + 16 + hsel*8 + (e-8)
// i.e. two contiguous 16-byte runs at (kk + hsel*8) and (+16).
__device__ __forceinline__ v16h load_frag(const _Float16* __restrict__ base,
                                          int ld, int row, int kk, int hsel) {
  Frag u;
  const _Float16* p = base + (size_t)row * ld + kk + hsel * 8;
  u.q[0] = *(const f4*)(p);
  u.q[1] = *(const f4*)(p + 16);
  return u.v;
}

// Same fragment assembled from fp32 data living in LDS (converted on the fly).
__device__ __forceinline__ v16h frag_from_lds_f32(const float* __restrict__ p0) {
  Frag u;
#pragma unroll
  for (int i = 0; i < 8; ++i) u.h[i] = (_Float16)p0[i];
#pragma unroll
  for (int i = 0; i < 8; ++i) u.h[8 + i] = (_Float16)p0[16 + i];
  return u.v;
}

__device__ __forceinline__ v8f wmma16(v16h a, v16h b, v8f c) {
  return __builtin_amdgcn_wmma_f32_16x16x32_f16(false, a, false, b,
                                                (short)0, c, false, false);
}

__device__ __forceinline__ float sigmoidf_(float x) {
  return 1.0f / (1.0f + __expf(-x));
}

// ---------------------------------------------------------------------------
// Embedding: x[b,s,:] = sum_c emb[seq[b,s,c], :]   (fp32 + f16 copy)
// ---------------------------------------------------------------------------
__global__ void embed_kernel(const int* __restrict__ seqs,
                             const float* __restrict__ emb,
                             float* __restrict__ x, _Float16* __restrict__ x16) {
  int bs = blockIdx.x;          // 0..BS_-1
  int d  = threadIdx.x;         // 0..255
  const int* row = seqs + (size_t)bs * C_;
  float acc = 0.f;
#pragma unroll 4
  for (int c = 0; c < C_; ++c) acc += emb[(size_t)row[c] * D_ + d];
  x[(size_t)bs * D_ + d]   = acc;
  x16[(size_t)bs * D_ + d] = (_Float16)acc;
}

__global__ void cast_f16_kernel(const float* __restrict__ in,
                                _Float16* __restrict__ out, int n) {
  int i = blockIdx.x * 256 + threadIdx.x;
  if (i < n) out[i] = (_Float16)in[i];
}

__global__ void zero_f16_kernel(_Float16* __restrict__ p, int n) {
  int i = blockIdx.x * 256 + threadIdx.x;
  if (i < n) p[i] = (_Float16)0.f;
}

// W_diff (200x200) -> padded f16 (208x224), zero-filled pad.
__global__ void wd_pad_kernel(const float* __restrict__ Wd,
                              _Float16* __restrict__ out) {
  int i = blockIdx.x * 256 + threadIdx.x;
  if (i >= MP_ * SP_) return;
  int r = i / SP_, c = i % SP_;
  out[i] = (r < S_ && c < S_) ? (_Float16)Wd[r * S_ + c] : (_Float16)0.f;
}

// alpha_cum = cumprod(1-betas); tsb[b] = time_scale[diffusion_t[b]]
__global__ void prep_kernel(const float* __restrict__ betas,
                            const int* __restrict__ dt,
                            const float* __restrict__ tscale,
                            float* __restrict__ alpha_cum,
                            float* __restrict__ tsb) {
  if (threadIdx.x == 0) {
    float a = 1.f;
    for (int i = 0; i < T_; ++i) { a *= (1.f - betas[i]); alpha_cum[i] = a; }
  }
  if (threadIdx.x < B_) tsb[threadIdx.x] = tscale[dt[threadIdx.x]];
}

// ---------------------------------------------------------------------------
// Generic WMMA GEMM: C[M][N] = A[M][K] * B[N][K]^T (+bias[N]) (*scale[batch])
// block = 128 threads (4 waves); each wave: 16 rows x 64 cols (4 WMMA tiles).
// The 64-row B panel for this block is DMA'd into LDS with one TDM
// tensor_load_to_lds (2D descriptor: tile = K x 64 rows, stride = ldb),
// so the K-loop's B traffic is ds_load while the A stream is software-
// pipelined global_load_b128 hidden behind the WMMAs.
// Requires: N multiple of 64; K multiple of 32, K <= 256; tiles in-bounds.
// ---------------------------------------------------------------------------
__global__ __launch_bounds__(128)
void wmma_gemm_kernel(const _Float16* __restrict__ A,
                      const _Float16* __restrict__ B,
                      float* __restrict__ C,
                      const float* __restrict__ bias,   // may be null
                      const float* __restrict__ scale,  // per-batch, may be null
                      int M, int N, int K,
                      int lda, int ldb, int ldc, int Mstore,
                      long sA, long sB, long sC) {
  __shared__ _Float16 Bs[64 * 256];    // staged B panel [64][K], up to 32 KB

  const int lane = threadIdx.x & 31;
  const int wave = threadIdx.x >> 5;
  const int hsel = lane >> 4;
  const int nl   = lane & 15;
  const int bz   = blockIdx.z;
  A += (size_t)bz * sA;  B += (size_t)bz * sB;  C += (size_t)bz * sC;

  const int m0 = blockIdx.x * 64 + wave * 16;
  const int n0 = blockIdx.y * 64;

  // ---- TDM: stage B[n0 .. n0+63][0..K) into LDS; one op issued by wave 0 ----
  if (threadIdx.x < 32) {
    unsigned lds_off = (unsigned)(unsigned long long)(const void*)&Bs[0];
    unsigned long long ga =
        (unsigned long long)(const void*)(B + (size_t)n0 * ldb);
    u32x4 g0;
    i32x8 g1;
    i32x4 g2 = (i32x4)0, g3 = (i32x4)0;
    i32x8 g4 = (i32x8)0;
    // D# group 0: count=1 (valid user descriptor), lds_addr, global_addr, type=2
    g0[0] = 1u;
    g0[1] = lds_off;
    g0[2] = (unsigned)ga;
    g0[3] = ((unsigned)(ga >> 32) & 0x01FFFFFFu) | (2u << 30);
    // D# group 1: wg_mask=0, data_size=1 (2B), dims/strides in elements
    g1[0] = (int)(1u << 16);                              // data_size = f16
    g1[1] = (int)(((unsigned)K & 0xFFFFu) << 16);         // tensor_dim0 lo16
    g1[2] = (int)((((unsigned)K >> 16) & 0xFFFFu) |
                  (((unsigned)N & 0xFFFFu) << 16));       // dim0 hi | dim1 lo
    g1[3] = (int)((((unsigned)N >> 16) & 0xFFFFu) |
                  (((unsigned)K & 0xFFFFu) << 16));       // dim1 hi | tile_dim0
    g1[4] = 64;                                           // tile_dim1=64, tile_dim2=0
    g1[5] = ldb;                                          // tensor_dim0_stride lo32
    g1[6] = 0;
    g1[7] = 0;
    __builtin_amdgcn_tensor_load_to_lds(g0, g1, g2, g3, g4, 0);
    __builtin_amdgcn_s_wait_tensorcnt(0);
  }
  __syncthreads();

  if (m0 < M) {
    const int arow = m0 + nl;
    v8f acc0 = {}, acc1 = {}, acc2 = {}, acc3 = {};

    // software-pipelined A stream: fetch kk+32 before computing kk
    v16h a = load_frag(A, lda, arow, 0, hsel);
    for (int kk = 0; kk < K; kk += 32) {
      v16h anext = a;
      if (kk + 32 < K) {
        __builtin_prefetch(A + (size_t)arow * lda + kk + 64, 0, 1);
        anext = load_frag(A, lda, arow, kk + 32, hsel);
      }
      v16h b0 = load_frag(Bs, K, nl,      kk, hsel);
      v16h b1 = load_frag(Bs, K, nl + 16, kk, hsel);
      v16h b2 = load_frag(Bs, K, nl + 32, kk, hsel);
      v16h b3 = load_frag(Bs, K, nl + 48, kk, hsel);
      acc0 = wmma16(a, b0, acc0);
      acc1 = wmma16(a, b1, acc1);
      acc2 = wmma16(a, b2, acc2);
      acc3 = wmma16(a, b3, acc3);
      a = anext;
    }

    const float sc = scale ? scale[bz] : 1.0f;
    const int mbase = m0 + hsel * 8;   // C/D layout: M = vgpr + 8*(lane>=16)
#pragma unroll
    for (int i = 0; i < 8; ++i) {
      int m = mbase + i;
      if (m < Mstore) {
        float* cr = C + (size_t)m * ldc + n0 + nl;
        cr[0]  = acc0[i] * sc + (bias ? bias[n0 + nl]      : 0.f);
        cr[16] = acc1[i] * sc + (bias ? bias[n0 + nl + 16] : 0.f);
        cr[32] = acc2[i] * sc + (bias ? bias[n0 + nl + 32] : 0.f);
        cr[48] = acc3[i] * sc + (bias ? bias[n0 + nl + 48] : 0.f);
      }
    }
  }
}

// ---------------------------------------------------------------------------
// Fused GRU scan. One workgroup (32 waves) on one WGP; h (64x256 fp32) lives
// in LDS. Each wave owns 2 "gate triples": a 16x16 tile of the hidden columns
// computed for r/z/n gates simultaneously (N, N+256, N+512 of hp), so the
// elementwise GRU update is wave-local. Whh fragment loads are software-
// pipelined across the K loop. Two barriers per timestep.
// ---------------------------------------------------------------------------
__global__ __launch_bounds__(1024)
void gru_rnn_kernel(const float* __restrict__ xp,      // [64][S][768]
                    const _Float16* __restrict__ Whh,  // [768][256] f16
                    const float* __restrict__ bhh,     // [768]
                    float* __restrict__ rnn,           // [64][S][256]
                    _Float16* __restrict__ rnn16) {    // [64][S][256]
  __shared__ float h[B_][D_];     // 64 KB of the WGP's 320 KB LDS
  __shared__ float bh[G3_];

  const int tid = threadIdx.x;
  for (int i = tid; i < B_ * D_; i += 1024) ((float*)h)[i] = 0.f;
  for (int i = tid; i < G3_; i += 1024) bh[i] = bhh[i];
  __syncthreads();

  const int lane = tid & 31;
  const int wave = tid >> 5;
  const int hsel = lane >> 4;
  const int nl   = lane & 15;

  for (int t = 0; t < S_; ++t) {
    float hnew[2][8];
#pragma unroll
    for (int p = 0; p < 2; ++p) {
      const int tri = wave * 2 + p;       // 0..63
      const int m0  = (tri >> 4) * 16;    // batch-row tile
      const int j0  = (tri & 15) * 16;    // hidden-column tile
      const float* hrow = &h[m0 + nl][0];

      v8f ar = {}, az = {}, an = {};
      v16h br = load_frag(Whh, D_, j0 + nl,       0, hsel);
      v16h bz = load_frag(Whh, D_, 256 + j0 + nl, 0, hsel);
      v16h bn = load_frag(Whh, D_, 512 + j0 + nl, 0, hsel);
      for (int kk = 0; kk < D_; kk += 32) {
        v16h a = frag_from_lds_f32(hrow + kk + hsel * 8);
        v16h nbr = br, nbz = bz, nbn = bn;
        if (kk + 32 < D_) {
          nbr = load_frag(Whh, D_, j0 + nl,       kk + 32, hsel);
          nbz = load_frag(Whh, D_, 256 + j0 + nl, kk + 32, hsel);
          nbn = load_frag(Whh, D_, 512 + j0 + nl, kk + 32, hsel);
        }
        ar = wmma16(a, br, ar);
        az = wmma16(a, bz, az);
        an = wmma16(a, bn, an);
        br = nbr; bz = nbz; bn = nbn;
      }

      const int j = j0 + nl;
#pragma unroll
      for (int i = 0; i < 8; ++i) {
        const int m = m0 + hsel * 8 + i;
        const float* xpp = xp + ((size_t)m * S_ + t) * G3_;
        float xr = xpp[j], xz = xpp[256 + j], xn = xpp[512 + j];
        float hr = ar[i] + bh[j];
        float hz = az[i] + bh[256 + j];
        float hn = an[i] + bh[512 + j];
        float rr = sigmoidf_(xr + hr);
        float zz = sigmoidf_(xz + hz);
        float nn = tanhf(xn + rr * hn);
        float hv = (1.f - zz) * nn + zz * h[m][j];
        hnew[p][i] = hv;
        size_t o = ((size_t)m * S_ + t) * D_ + j;
        rnn[o]   = hv;
        rnn16[o] = (_Float16)hv;
      }
    }
    __syncthreads();                      // everyone done reading old h
#pragma unroll
    for (int p = 0; p < 2; ++p) {
      const int tri = wave * 2 + p;
      const int m0  = (tri >> 4) * 16;
      const int j   = (tri & 15) * 16 + nl;
#pragma unroll
      for (int i = 0; i < 8; ++i) h[m0 + hsel * 8 + i][j] = hnew[p][i];
    }
    __syncthreads();                      // new h visible for next step
  }
}

// ---------------------------------------------------------------------------
// Attention MLP + blend + aligned + x_noisy (written K-transposed in f16 for
// the diffusion GEMM). One block per (s, b).
// ---------------------------------------------------------------------------
__global__ void attn_blend_kernel(const float* __restrict__ x1,
                                  const float* __restrict__ wprev,   // [64][S][256]
                                  const float* __restrict__ w1W, const float* __restrict__ w1b,
                                  const float* __restrict__ w2W, const float* __restrict__ w2b,
                                  const int* __restrict__ dt,
                                  const float* __restrict__ alpha_cum,
                                  const float* __restrict__ noise,
                                  float* __restrict__ aligned,
                                  _Float16* __restrict__ xnT) {      // [64][256][224]
  const int s = blockIdx.x, b = blockIdx.y, d = threadIdx.x;
  __shared__ float cat[2 * D_];
  __shared__ float hv[64];
  __shared__ float lg[2];

  const float* ek = x1 + (size_t)b * S_ * D_;   // x[b,0,:]
  cat[d] = ek[d];
  float a0 = 1.f, a1 = 0.f;
  if (s > 0) {
    cat[D_ + d] = wprev[((size_t)b * S_ + (s - 1)) * D_ + d];
    __syncthreads();
    if (d < 64) {
      float acc = w1b[d];
      const float* wr = w1W + d * (2 * D_);
      for (int e = 0; e < 2 * D_; ++e) acc += wr[e] * cat[e];
      hv[d] = tanhf(acc);
    }
    __syncthreads();
    if (d < 2) {
      float acc = w2b[d];
      const float* wr = w2W + d * 64;
      for (int f = 0; f < 64; ++f) acc += wr[f] * hv[f];
      lg[d] = acc;
    }
    __syncthreads();
    float mx = fmaxf(lg[0], lg[1]);
    float e0 = __expf(lg[0] - mx), e1 = __expf(lg[1] - mx);
    float inv = 1.f / (e0 + e1);
    a0 = e0 * inv;  a1 = e1 * inv;
  }
  float al = (s == 0) ? ek[d] : (cat[d] * a0 + cat[D_ + d] * a1);
  aligned[((size_t)b * S_ + s) * D_ + d] = al;

  float ac = alpha_cum[dt[b]];
  float xn = al * sqrtf(ac) + noise[((size_t)b * S_ + s) * D_ + d] * sqrtf(1.f - ac);
  xnT[((size_t)b * D_ + d) * SP_ + s] = (_Float16)xn;
}

// gen_x = aligned + noise - predicted_noise  (f16 for the xp2 GEMM)
__global__ void genx_kernel(const float* __restrict__ aligned,
                            const float* __restrict__ noise,
                            const float* __restrict__ pnoise,
                            _Float16* __restrict__ gx16, int n) {
  int i = blockIdx.x * 256 + threadIdx.x;
  if (i < n) gx16[i] = (_Float16)(aligned[i] + noise[i] - pnoise[i]);
}

// ---------------------------------------------------------------------------
// Attention pool: softmax over S of rnn@wl, weighted feature, 2-way head.
// grid = (B, 2): which=0 -> pred, which=1 -> gen_pred.
// ---------------------------------------------------------------------------
__global__ void pool_kernel(const float* __restrict__ rnn1,
                            const float* __restrict__ rnn2,
                            const float* __restrict__ wlW, const float* __restrict__ wlb,
                            const float* __restrict__ outW, const float* __restrict__ outb,
                            float* __restrict__ out) {
  const int b = blockIdx.x, which = blockIdx.y;
  const float* rnn = (which == 0 ? rnn1 : rnn2) + (size_t)b * S_ * D_;
  __shared__ float att[S_];
  __shared__ float feat[D_];
  __shared__ float inv_s;
  const int tid = threadIdx.x;

  if (tid < S_) {
    float acc = wlb[0];
    const float* r = rnn + (size_t)tid * D_;
    for (int dd = 0; dd < D_; ++dd) acc += r[dd] * wlW[dd];
    att[tid] = acc;
  }
  __syncthreads();
  if (tid == 0) {
    float mx = -1e30f;
    for (int s = 0; s < S_; ++s) mx = fmaxf(mx, att[s]);
    float sum = 0.f;
    for (int s = 0; s < S_; ++s) { att[s] = __expf(att[s] - mx); sum += att[s]; }
    inv_s = 1.f / sum;
  }
  __syncthreads();
  {
    float acc = 0.f;
    for (int s = 0; s < S_; ++s) acc += att[s] * rnn[(size_t)s * D_ + tid];
    feat[tid] = acc * inv_s;
  }
  __syncthreads();
  if (tid < 2) {
    float acc = outb[tid];
    for (int dd = 0; dd < D_; ++dd) acc += outW[tid * D_ + dd] * feat[dd];
    out[which * (B_ * 2) + b * 2 + tid] = acc;
  }
}

// ---------------------------------------------------------------------------
extern "C" void kernel_launch(void* const* d_in, const int* in_sizes, int n_in,
                              void* d_out, int out_size, void* d_ws, size_t ws_size,
                              hipStream_t stream) {
  const int*   seqs  = (const int*)d_in[0];
  const int*   dt    = (const int*)d_in[5];
  const float* emb   = (const float*)d_in[6];
  const float* Wih   = (const float*)d_in[7];
  const float* Whh   = (const float*)d_in[8];
  const float* bih   = (const float*)d_in[9];
  const float* bhh   = (const float*)d_in[10];
  const float* whkW  = (const float*)d_in[11];
  const float* whkb  = (const float*)d_in[12];
  const float* w1W   = (const float*)d_in[13];
  const float* w1b   = (const float*)d_in[14];
  const float* w2W   = (const float*)d_in[15];
  const float* w2b   = (const float*)d_in[16];
  const float* wlW   = (const float*)d_in[17];
  const float* wlb   = (const float*)d_in[18];
  const float* outW  = (const float*)d_in[19];
  const float* outb  = (const float*)d_in[20];
  const float* betas = (const float*)d_in[21];
  const float* Wd    = (const float*)d_in[22];
  const float* tsc   = (const float*)d_in[23];
  const float* noise = (const float*)d_in[24];
  float* out = (float*)d_out;
  float* pnoise_out = out + 2 * (B_ * 2);           // predicted_noise region
  float* nnoise_out = pnoise_out + (size_t)BS_ * D_;

  // ---- workspace carve (≈133 MB) ----
  char* wp = (char*)d_ws;
  auto alloc = [&](size_t bytes) {
    char* p = wp; wp += (bytes + 255) & ~(size_t)255; return p;
  };
  float*    x1     = (float*)   alloc((size_t)BS_ * D_ * 4);
  _Float16* x1_16  = (_Float16*)alloc((size_t)BS_ * D_ * 2); // later reused as rnn2 f16 sink
  _Float16* Wih16  = (_Float16*)alloc((size_t)G3_ * D_ * 2);
  _Float16* Whh16  = (_Float16*)alloc((size_t)G3_ * D_ * 2);
  _Float16* whk16  = (_Float16*)alloc((size_t)D_ * D_ * 2);
  _Float16* Wd16   = (_Float16*)alloc((size_t)MP_ * SP_ * 2);
  float*    xp     = (float*)   alloc((size_t)BS_ * G3_ * 4); // xp1 then xp2
  float*    rnn1   = (float*)   alloc((size_t)BS_ * D_ * 4);
  float*    rnn2   = (float*)   alloc((size_t)BS_ * D_ * 4);
  _Float16* rnn116 = (_Float16*)alloc((size_t)BS_ * D_ * 2);
  float*    wprev  = (float*)   alloc((size_t)BS_ * D_ * 4);
  float*    alig   = (float*)   alloc((size_t)BS_ * D_ * 4);
  _Float16* xnT    = (_Float16*)alloc((size_t)B_ * D_ * SP_ * 2);
  _Float16* gx16   = (_Float16*)alloc((size_t)BS_ * D_ * 2);
  float*    acum   = (float*)   alloc((size_t)T_ * 4);
  float*    tsb    = (float*)   alloc((size_t)B_ * 4);

  // 1) embedding (fp32 + f16)
  embed_kernel<<<BS_, D_, 0, stream>>>(seqs, emb, x1, x1_16);

  // 2) weight conversions
  cast_f16_kernel<<<(G3_ * D_ + 255) / 256, 256, 0, stream>>>(Wih, Wih16, G3_ * D_);
  cast_f16_kernel<<<(G3_ * D_ + 255) / 256, 256, 0, stream>>>(Whh, Whh16, G3_ * D_);
  cast_f16_kernel<<<(D_ * D_ + 255) / 256, 256, 0, stream>>>(whkW, whk16, D_ * D_);
  wd_pad_kernel<<<(MP_ * SP_ + 255) / 256, 256, 0, stream>>>(Wd, Wd16);
  prep_kernel<<<1, 64, 0, stream>>>(betas, dt, tsc, acum, tsb);
  zero_f16_kernel<<<(B_ * D_ * SP_ + 255) / 256, 256, 0, stream>>>(xnT, B_ * D_ * SP_);

  // 3) xp1 = x @ Wih^T + bih  (12800x768, K=256)
  wmma_gemm_kernel<<<dim3(BS_ / 64, G3_ / 64, 1), 128, 0, stream>>>(
      x1_16, Wih16, xp, bih, nullptr,
      BS_, G3_, D_, D_, D_, G3_, BS_, 0, 0, 0);

  // 4) GRU scan 1
  gru_rnn_kernel<<<1, 1024, 0, stream>>>(xp, Whh16, bhh, rnn1, rnn116);

  // 5) w_prev = rnn1 @ whk_W^T + whk_b  (12800x256, K=256)
  wmma_gemm_kernel<<<dim3(BS_ / 64, D_ / 64, 1), 128, 0, stream>>>(
      rnn116, whk16, wprev, whkb, nullptr,
      BS_, D_, D_, D_, D_, D_, BS_, 0, 0, 0);

  // 6) attention blend + aligned + x_noisy^T (f16)
  attn_blend_kernel<<<dim3(S_, B_), D_, 0, stream>>>(
      x1, wprev, w1W, w1b, w2W, w2b, dt, acum, noise, alig, xnT);

  // 7) predicted_noise[b] = (W_diff @ x_noisy[b]) * time_scale[t_b]
  //    batched GEMM: M=208(store 200), N=256, K=224
  wmma_gemm_kernel<<<dim3((MP_ + 63) / 64, D_ / 64, B_), 128, 0, stream>>>(
      Wd16, xnT, pnoise_out, nullptr, tsb,
      MP_, D_, SP_, SP_, SP_, D_, S_,
      0, (long)D_ * SP_, (long)S_ * D_);

  // 8) gen_x = aligned + noise - predicted_noise (f16)
  genx_kernel<<<(BS_ * D_ + 255) / 256, 256, 0, stream>>>(
      alig, noise, pnoise_out, gx16, BS_ * D_);

  // 9) xp2 = gen_x @ Wih^T + bih (reuse xp buffer)
  wmma_gemm_kernel<<<dim3(BS_ / 64, G3_ / 64, 1), 128, 0, stream>>>(
      gx16, Wih16, xp, bih, nullptr,
      BS_, G3_, D_, D_, D_, G3_, BS_, 0, 0, 0);

  // 10) GRU scan 2 (f16 sink reuses x1_16; x1 fp32 no longer needed after 6)
  gru_rnn_kernel<<<1, 1024, 0, stream>>>(xp, Whh16, bhh, rnn2, x1_16);

  // 11) attention pools -> pred / gen_pred
  pool_kernel<<<dim3(B_, 2), D_, 0, stream>>>(rnn1, rnn2, wlW, wlb, outW, outb, out);

  // 12) normal_noise pass-through
  (void)hipMemcpyAsync(nnoise_out, noise, (size_t)BS_ * D_ * sizeof(float),
                       hipMemcpyDeviceToDevice, stream);
}
